// VectorQuantizer_9869834846740
// MI455X (gfx1250) — matrix operand order repaired and checked by reading, not compiled, not run on hardware
//
#include <hip/hip_runtime.h>
#include <hip/hip_bf16.h>
#include <math.h>

typedef float  v8f  __attribute__((ext_vector_type(8)));
typedef float  v4f  __attribute__((ext_vector_type(4)));
typedef __bf16 v16bf __attribute__((ext_vector_type(16)));
typedef __bf16 v8bf  __attribute__((ext_vector_type(8)));
typedef __bf16 v4bf  __attribute__((ext_vector_type(4)));

#define N_TOK   32768
#define K_EMB   4096
#define D_EMB   256
#define TILE_ROWS 64

// d_out flat layout (floats), tuple order: quantized_st, perplexity, encodings, indices, loss
#define OFF_Q    ((size_t)0)
#define OFF_PERP ((size_t)8388608)
#define OFF_ENC  ((size_t)8388609)
#define OFF_IDX  ((size_t)142606337)
#define OFF_LOSS ((size_t)142639105)
#define OUT_TOTAL ((size_t)142639106)

#define SHUF16(a, b) __builtin_shufflevector(a, b, 0,1,2,3,4,5,6,7,8,9,10,11,12,13,14,15)

// ---------------------------------------------------------------- zero fill (one-hot region + ws counters)
__global__ __launch_bounds__(256) void vq_zero_kernel(float* __restrict__ out,
                                                      float* __restrict__ wsf) {
    // encodings region [OFF_ENC, OFF_IDX): peel 3 head, 33,554,431 float4s, 1 tail
    float* p4 = out + OFF_ENC + 3;                 // 16B-aligned
    const size_t n4 = 33554431;
    size_t i = (size_t)blockIdx.x * blockDim.x + threadIdx.x;
    const size_t stride = (size_t)gridDim.x * blockDim.x;
    v4f z = {0.f, 0.f, 0.f, 0.f};
    for (size_t j = i; j < n4; j += stride)
        __builtin_nontemporal_store(z, (v4f*)(p4 + 4 * j));
    if (i < 3) out[OFF_ENC + i] = 0.0f;
    else if (i == 3) out[OFF_IDX - 1] = 0.0f;
    // ws dwords [4096 .. 8192]: counts[4096] + lsum
    if (i < 4097) wsf[4096 + i] = 0.0f;
}

// ---------------------------------------------------------------- split W into bf16 hi/lo
__global__ __launch_bounds__(256) void vq_wprep_kernel(const float* __restrict__ W,
                                                       __bf16* __restrict__ whi,
                                                       __bf16* __restrict__ wlo) {
    size_t t = (size_t)blockIdx.x * 256 + threadIdx.x;   // over K*D/4 float4s
    float4 w = *(const float4*)(W + 4 * t);
    __bf16 h0 = (__bf16)w.x, h1 = (__bf16)w.y, h2 = (__bf16)w.z, h3 = (__bf16)w.w;
    __bf16 l0 = (__bf16)(w.x - (float)h0);
    __bf16 l1 = (__bf16)(w.y - (float)h1);
    __bf16 l2 = (__bf16)(w.z - (float)h2);
    __bf16 l3 = (__bf16)(w.w - (float)h3);
    v4bf hv = {h0, h1, h2, h3};
    v4bf lv = {l0, l1, l2, l3};
    *(v4bf*)(whi + 4 * t) = hv;
    *(v4bf*)(wlo + 4 * t) = lv;
}

// ---------------------------------------------------------------- ||e_k||^2 (full f32)
__global__ __launch_bounds__(256) void vq_enorm_kernel(const float* __restrict__ W,
                                                       float* __restrict__ enorm) {
    int gwave = (int)((blockIdx.x * blockDim.x + threadIdx.x) >> 5);
    int lane  = threadIdx.x & 31;
    if (gwave >= K_EMB) return;
    const float* row = W + (size_t)gwave * D_EMB + lane * 8;
    float4 a = *(const float4*)(row);
    float4 b = *(const float4*)(row + 4);
    float s = a.x*a.x + a.y*a.y + a.z*a.z + a.w*a.w
            + b.x*b.x + b.y*b.y + b.z*b.z + b.w*b.w;
    #pragma unroll
    for (int off = 16; off > 0; off >>= 1)
        s += __shfl_xor(s, off, 32);
    if (lane == 0) enorm[gwave] = s;
}

// ---------------------------------------------------------------- main WMMA argmin (bf16x3)
// Block: 128 threads (4 waves), 64 rows. Wave w: row-pair (w&1) -> 32 rows, K-half (w>>1).
// Two 16-row C tiles per wave share every B fragment (halves B-operand traffic per MAC).
// LDS: X tile split hi/lo bf16, 16-byte-chunk xor swizzle: phys_chunk = chunk ^ (row&15).
__global__ __launch_bounds__(128) void vq_argmin_kernel(const float* __restrict__ X,
                                                        const __bf16* __restrict__ Whi,
                                                        const __bf16* __restrict__ Wlo,
                                                        const float* __restrict__ enorm,
                                                        float* __restrict__ outIdxF) {
    __shared__ __bf16 xhi[TILE_ROWS * D_EMB];   // 32 KB
    __shared__ __bf16 xlo[TILE_ROWS * D_EMB];   // 32 KB
    const int t    = threadIdx.x;
    const int row0 = blockIdx.x * TILE_ROWS;

    // Stage + split X tile: 64 rows x 64 float4
    for (int i = t; i < TILE_ROWS * 64; i += 128) {
        int r   = i >> 6;
        int col = (i & 63) << 2;                 // multiple of 4
        float4 v = *(const float4*)(X + (size_t)(row0 + r) * D_EMB + col);
        __bf16 h0 = (__bf16)v.x, h1 = (__bf16)v.y, h2 = (__bf16)v.z, h3 = (__bf16)v.w;
        __bf16 l0 = (__bf16)(v.x - (float)h0);
        __bf16 l1 = (__bf16)(v.y - (float)h1);
        __bf16 l2 = (__bf16)(v.z - (float)h2);
        __bf16 l3 = (__bf16)(v.w - (float)h3);
        int pc = ((((col >> 3) ^ (r & 15)) << 3) | (col & 7)) + r * D_EMB;
        v4bf hv = {h0, h1, h2, h3};
        v4bf lv = {l0, l1, l2, l3};
        *(v4bf*)(&xhi[pc]) = hv;
        *(v4bf*)(&xlo[pc]) = lv;
    }
    __syncthreads();

    const int wave = t >> 5;
    const int lane = t & 31;
    const int half = lane >> 4;   // fragment K-slice select
    const int lrow = lane & 15;   // A row (M), B/C column (N)
    const int rp   = wave & 1;    // row-pair in block (32 rows)
    const int kh   = wave >> 1;   // K half (0: cols 0..2047, 1: 2048..4095)
    const __bf16* aHi0 = &xhi[(rp * 32 + lrow) * D_EMB];
    const __bf16* aLo0 = &xlo[(rp * 32 + lrow) * D_EMB];
    const __bf16* aHi1 = aHi0 + 16 * D_EMB;
    const __bf16* aLo1 = aLo0 + 16 * D_EMB;

    float bv[16];
    int   bi[16];
    #pragma unroll
    for (int j = 0; j < 16; ++j) { bv[j] = -3.0e38f; bi[j] = 0; }

    const int ktBeg = kh * 128;
    for (int kt = ktBeg; kt < ktBeg + 128; ++kt) {
        const int k0   = kt << 4;
        const int kcol = k0 + lrow;
        // B fragment: two 8-bf16 chunks of one W row; K [8h,8h+8) and [16+8h,16+8h+8)
        const __bf16* bHi = Whi + (size_t)kcol * D_EMB + 8 * half;
        const __bf16* bLo = Wlo + (size_t)kcol * D_EMB + 8 * half;
        v8f c0 = {};
        v8f c1 = {};
        #pragma unroll
        for (int s = 0; s < 8; ++s) {           // d0 = 32*s
            // A fragment chunks (8 bf16 = 16B each): K [8h,8h+8) and [16+8h,16+8h+8) rel. d0
            int ch0 = ((4 * s + half)     ^ lrow) << 3;
            int ch1 = ((4 * s + 2 + half) ^ lrow) << 3;
            v8bf ah0a = *(const v8bf*)(aHi0 + ch0);
            v8bf ah0b = *(const v8bf*)(aHi0 + ch1);
            v8bf ah1a = *(const v8bf*)(aHi1 + ch0);
            v8bf ah1b = *(const v8bf*)(aHi1 + ch1);
            v8bf al0a = *(const v8bf*)(aLo0 + ch0);
            v8bf al0b = *(const v8bf*)(aLo0 + ch1);
            v8bf al1a = *(const v8bf*)(aLo1 + ch0);
            v8bf al1b = *(const v8bf*)(aLo1 + ch1);
            v16bf ahi0 = SHUF16(ah0a, ah0b);
            v16bf ahi1 = SHUF16(ah1a, ah1b);
            v16bf alo0 = SHUF16(al0a, al0b);
            v16bf alo1 = SHUF16(al1a, al1b);
            v8bf bha = *(const v8bf*)(bHi + 32 * s);
            v8bf bhb = *(const v8bf*)(bHi + 32 * s + 16);
            v8bf bla = *(const v8bf*)(bLo + 32 * s);
            v8bf blb = *(const v8bf*)(bLo + 32 * s + 16);
            v16bf bhi = SHUF16(bha, bhb);
            v16bf blo = SHUF16(bla, blb);
            // bf16x3: hi*hi + hi*lo + lo*hi, f32 accumulate; B shared by both row tiles
            c0 = __builtin_amdgcn_wmma_f32_16x16x32_bf16(false, ahi0, false, bhi, (short)0, c0, false, false);
            c1 = __builtin_amdgcn_wmma_f32_16x16x32_bf16(false, ahi1, false, bhi, (short)0, c1, false, false);
            c0 = __builtin_amdgcn_wmma_f32_16x16x32_bf16(false, ahi0, false, blo, (short)0, c0, false, false);
            c1 = __builtin_amdgcn_wmma_f32_16x16x32_bf16(false, ahi1, false, blo, (short)0, c1, false, false);
            c0 = __builtin_amdgcn_wmma_f32_16x16x32_bf16(false, alo0, false, bhi, (short)0, c0, false, false);
            c1 = __builtin_amdgcn_wmma_f32_16x16x32_bf16(false, alo1, false, bhi, (short)0, c1, false, false);
        }
        const float sc = 0.5f * enorm[kcol];
        #pragma unroll
        for (int j = 0; j < 8; ++j) {
            float v0 = c0[j] - sc;
            if (v0 > bv[j])     { bv[j]     = v0; bi[j]     = kcol; }
            float v1 = c1[j] - sc;
            if (v1 > bv[8 + j]) { bv[8 + j] = v1; bi[8 + j] = kcol; }
        }
    }

    // Reduce across the 16 lanes of each half (half 0 holds rows +0..7, half 1 rows +8..15)
    #pragma unroll
    for (int off = 1; off < 16; off <<= 1) {
        #pragma unroll
        for (int j = 0; j < 16; ++j) {
            float ov = __shfl_xor(bv[j], off, 32);
            int   oi = __shfl_xor(bi[j], off, 32);
            if (ov > bv[j] || (ov == bv[j] && oi < bi[j])) { bv[j] = ov; bi[j] = oi; }
        }
    }

    __syncthreads();                       // done reading X tile; reuse LDS for merge
    float* mval = (float*)xhi;             // [kh*64 + rowInBlock]
    int*   midx = (int*)xhi + 128;
    if (lrow == 0) {
        #pragma unroll
        for (int ti = 0; ti < 2; ++ti) {
            #pragma unroll
            for (int j = 0; j < 8; ++j) {
                int rowInBlock = rp * 32 + ti * 16 + half * 8 + j;
                mval[kh * 64 + rowInBlock] = bv[ti * 8 + j];
                midx[kh * 64 + rowInBlock] = bi[ti * 8 + j];
            }
        }
    }
    __syncthreads();
    if (t < 64) {
        float v0 = mval[t], v1 = mval[64 + t];
        int   i0 = midx[t], i1 = midx[64 + t];
        int best = (v1 > v0 || (v1 == v0 && i1 < i0)) ? i1 : i0;
        outIdxF[row0 + t] = (float)best;
    }
}

// ---------------------------------------------------------------- gather + ST + loss + counts
__global__ __launch_bounds__(256) void vq_gather_kernel(const float* __restrict__ X,
                                                        const float* __restrict__ W,
                                                        const float* __restrict__ idxF,
                                                        float* __restrict__ outQ,
                                                        float* __restrict__ lsum,
                                                        unsigned* __restrict__ counts) {
    __shared__ float red[256];
    size_t t  = (size_t)blockIdx.x * 256 + threadIdx.x;   // over N*64 float4 units
    int n  = (int)(t >> 6);
    int c4 = (int)(t & 63) << 2;
    int idx = (int)idxF[n];
    float4 x = *(const float4*)(X + (size_t)n * D_EMB + c4);
    float4 q = *(const float4*)(W + (size_t)idx * D_EMB + c4);
    float4 d = make_float4(q.x - x.x, q.y - x.y, q.z - x.z, q.w - x.w);
    v4f st = {x.x + d.x, x.y + d.y, x.z + d.z, x.w + d.w};
    __builtin_nontemporal_store(st, (v4f*)(outQ + (size_t)n * D_EMB + c4));
    if (c4 == 0) atomicAdd(&counts[idx], 1u);
    float ls = d.x*d.x + d.y*d.y + d.z*d.z + d.w*d.w;
    red[threadIdx.x] = ls;
    __syncthreads();
    #pragma unroll
    for (int s = 128; s > 0; s >>= 1) {
        if ((int)threadIdx.x < s) red[threadIdx.x] += red[threadIdx.x + s];
        __syncthreads();
    }
    if (threadIdx.x == 0) atomicAdd(lsum, red[0]);
}

// ---------------------------------------------------------------- one-hot scatter
__global__ __launch_bounds__(256) void vq_scatter_kernel(const float* __restrict__ idxF,
                                                         float* __restrict__ enc) {
    int n = blockIdx.x * 256 + threadIdx.x;
    if (n >= N_TOK) return;
    int idx = (int)idxF[n];
    __builtin_nontemporal_store(1.0f, enc + (size_t)n * K_EMB + idx);
}

// ---------------------------------------------------------------- perplexity + loss
__global__ __launch_bounds__(256) void vq_final_kernel(const unsigned* __restrict__ counts,
                                                       const float* __restrict__ lsum,
                                                       float* __restrict__ outPerp,
                                                       float* __restrict__ outLoss) {
    __shared__ float red[256];
    float s = 0.0f;
    for (int k = threadIdx.x; k < K_EMB; k += 256) {
        float p = (float)counts[k] * (1.0f / (float)N_TOK);
        s += p * logf(p + 1e-10f);
    }
    red[threadIdx.x] = s;
    __syncthreads();
    #pragma unroll
    for (int st = 128; st > 0; st >>= 1) {
        if ((int)threadIdx.x < st) red[threadIdx.x] += red[threadIdx.x + st];
        __syncthreads();
    }
    if (threadIdx.x == 0) {
        *outPerp = expf(-red[0]);
        *outLoss = 1.25f * (*lsum) * (1.0f / ((float)N_TOK * (float)D_EMB));
    }
}

extern "C" void kernel_launch(void* const* d_in, const int* in_sizes, int n_in,
                              void* d_out, int out_size, void* d_ws, size_t ws_size,
                              hipStream_t stream) {
    const float* X = (const float*)d_in[0];   // [N, D]
    const float* W = (const float*)d_in[1];   // [K, D]
    float* out = (float*)d_out;
    float* wsf = (float*)d_ws;

    float*    enorm  = wsf;                                   // dwords 0..4095
    unsigned* counts = (unsigned*)(wsf + 4096);               // dwords 4096..8191
    float*    lsum   = wsf + 8192;                            // dword 8192
    __bf16*   whi    = (__bf16*)((char*)d_ws + 65536);        // 2 MB
    __bf16*   wlo    = whi + (size_t)K_EMB * D_EMB;           // 2 MB

    vq_zero_kernel   <<<2048, 256, 0, stream>>>(out, wsf);
    vq_wprep_kernel  <<<(K_EMB * D_EMB / 4) / 256, 256, 0, stream>>>(W, whi, wlo);
    vq_enorm_kernel  <<<K_EMB / 8, 256, 0, stream>>>(W, enorm);
    vq_argmin_kernel <<<N_TOK / TILE_ROWS, 128, 0, stream>>>(X, whi, wlo, enorm,
                                                             out + OFF_IDX);
    vq_gather_kernel <<<N_TOK * 64 / 256, 256, 0, stream>>>(X, W, out + OFF_IDX,
                                                            out + OFF_Q, lsum, counts);
    vq_scatter_kernel<<<N_TOK / 256, 256, 0, stream>>>(out + OFF_IDX, out + OFF_ENC);
    vq_final_kernel  <<<1, 256, 0, stream>>>(counts, lsum, out + OFF_PERP, out + OFF_LOSS);
}
